// FusionMamba_2104533975588
// MI455X (gfx1250) — matrix operand order repaired and checked by reading, not compiled, not run on hardware
//
#include <hip/hip_runtime.h>
#include <hip/hip_bf16.h>
#include <stdint.h>
#include <stddef.h>

// ---------------------------------------------------------------------------
// FusionMamba on gfx1250 (MI455X, wave32, WMMA).
//
// All dense projections run through v_wmma_f32_16x16x32_f16 (f16 A/B, f32 acc).
// Weights are pre-packed once per launch into the per-lane CDNA5 A-fragment
// layout (f16, zero-padded rows), so the GEMM K-loop is pure b128 loads +
// WMMA. The K-loop is software-pipelined via manual unroll-by-2 with
// alternating fragment buffer sets (no rotation copies): loads for step tk+1
// are in flight while step tk's 4 WMMAs execute.
// The sequential selective scan keeps the 16-wide SSM state in registers
// (one lane per inner channel) and fuses softplus/exp/dA/dBu/D-skip/SiLU-gate,
// so the (B,256,4096,16) dA/dBu tensors are never materialized.
// ---------------------------------------------------------------------------

typedef __attribute__((ext_vector_type(16))) _Float16 v16h;
typedef __attribute__((ext_vector_type(8)))  float    v8f;

#define BB    2
#define CC    128
#define LL    4096          // 64*64
#define DI    256           // EXPAND * C
#define DS    16            // D_STATE
#define DTR   8             // DT_RANK
#define XROWS (DTR + 2*DS)  // 40
#define NT    4             // l-tiles (of 16) per wave in the GEMM

__device__ __forceinline__ float sigmoidf_(float x) { return 1.f / (1.f + __expf(-x)); }

// ---------------------------------------------------------------------------
// Kernel 0: pack a weight matrix W (E x Dd, f32) into WMMA A-fragment order:
//   Apack[((te*nk + tk)*32 + lane)*16 + i]  (f16), Ep = ceil16(E), nk = Dd/32
// Contiguous k order used here (16x32 MxK; B loads match):
//   lane<16 : row M = te*16+lane,     K = tk*32 + {0..15}
//   lane>=16: row M = te*16+lane-16,  K = tk*32 + {16..31}
// Rows >= E are zero-filled, so the GEMM needs no bounds check on loads.
// ---------------------------------------------------------------------------
__global__ __launch_bounds__(256) void pack_w_kernel(
    const float* __restrict__ W, _Float16* __restrict__ Apack, int E, int Dd)
{
  const int Ep = (E + 15) & ~15;
  const size_t total = (size_t)Ep * Dd;
  const size_t idx = (size_t)blockIdx.x * 256 + threadIdx.x;
  if (idx >= total) return;

  const int i    = (int)(idx & 15);
  const int lane = (int)((idx >> 4) & 31);
  const size_t tt = idx >> 9;          // te*nk + tk
  const int nk = Dd >> 5;
  const int tk = (int)(tt % nk);
  const int te = (int)(tt / nk);
  const int half = lane >> 4;
  const int l16  = lane & 15;
  const int row = te * 16 + l16;
  const int k   = tk * 32 + half * 16 + i;

  Apack[idx] = (row < E) ? (_Float16)W[(size_t)row * Dd + k] : (_Float16)0.f;
}

// ---------------------------------------------------------------------------
// Kernel 1: layernorm over channels, emit f16 activations in (b, l, d) order
// so WMMA "B" fragments are contiguous 32-byte loads.
// ---------------------------------------------------------------------------
__global__ __launch_bounds__(CC) void ln_f16_kernel(
    const float* __restrict__ x,   // (B, C, L)
    const float* __restrict__ w,   // (C)
    const float* __restrict__ b,   // (C)
    _Float16* __restrict__ out)    // (B, L, C) f16
{
  const int l = blockIdx.x;
  const int bb = blockIdx.y;
  const int c = threadIdx.x;

  float v = x[((size_t)bb * CC + c) * LL + l];

  __shared__ float s1[CC];
  __shared__ float s2[CC];
  s1[c] = v;
  s2[c] = v * v;
  __syncthreads();
  for (int off = CC / 2; off > 0; off >>= 1) {
    if (c < off) { s1[c] += s1[c + off]; s2[c] += s2[c + off]; }
    __syncthreads();
  }
  const float mu  = s1[0] * (1.f / CC);
  const float var = s2[0] * (1.f / CC) - mu * mu;
  const float inv = rsqrtf(var + 1e-5f);

  out[((size_t)bb * LL + l) * CC + c] = (_Float16)((v - mu) * inv * w[c] + b[c]);
}

// ---------------------------------------------------------------------------
// Kernel 2: WMMA GEMM on packed weights (pipelined, alternating buffers):
//   Out[b, e, l] = sum_d W[e, d] * X[b, l, d]
//   Apack: packed f16 A fragments (see pack_w_kernel), Ep rows
//   X:     (B, L, Dd) f16
//   Out:   (B, E, L) f32
// One wave per 16(M=e) x 64(N=l) strip: per K-step, one A fragment (32B load)
// feeds 4 v_wmma_f32_16x16x32_f16. nk = Dd/32 is even for all shapes here,
// so the K-loop is unrolled by 2 with two fragment buffer sets: no rotation
// copies, and loads stay a full step ahead of the WMMAs that consume them.
// D layout (16x16 f32): VGPR r, lane<16 -> (M=r, N=lane);
//                       lane>=16 -> (M=r+8, N=lane-16).
// ---------------------------------------------------------------------------
__device__ __forceinline__ v16h gemm_load_a(const _Float16* ap, int tk) {
  return *(const v16h*)(ap + (size_t)tk * 32 * 16);
}
__device__ __forceinline__ void gemm_load_b(v16h bf[NT], const _Float16* xbase,
                                            int tk, int Dd) {
#pragma unroll
  for (int t = 0; t < NT; ++t)
    bf[t] = *(const v16h*)(xbase + (size_t)tk * 32 + (size_t)t * 16 * Dd);
}
__device__ __forceinline__ void gemm_wmma4(v8f acc[NT], v16h a, const v16h bf[NT]) {
#pragma unroll
  for (int t = 0; t < NT; ++t)
    acc[t] = __builtin_amdgcn_wmma_f32_16x16x32_f16(
        /*neg_a=*/false, a, /*neg_b=*/false, bf[t],
        /*c_mod=*/(short)0, acc[t], /*reuse_a=*/false, /*reuse_b=*/false);
}

__global__ __launch_bounds__(32) void wmma_gemm_packed_kernel(
    const _Float16* __restrict__ Apack,
    const _Float16* __restrict__ X,
    float* __restrict__ Out,
    int E, int Dd)
{
  const int lane = threadIdx.x;
  const int half = lane >> 4;
  const int l16  = lane & 15;
  const int l0 = blockIdx.x * (16 * NT);
  const int te = blockIdx.y;
  const int b  = blockIdx.z;
  const int nk = Dd >> 5;   // even (4 or 8) for all shapes in this model

  const _Float16* __restrict__ Xb = X + (size_t)b * LL * Dd;
  const _Float16* __restrict__ ap =
      Apack + ((size_t)te * nk * 32 + lane) * 16;
  const _Float16* __restrict__ xbase =
      Xb + (size_t)(l0 + l16) * Dd + half * 16;

  v8f acc[NT] = {};
  v16h a0, a1, b0[NT], b1[NT];

  // Prologue: fragments for tk = 0 into set 0.
  a0 = gemm_load_a(ap, 0);
  gemm_load_b(b0, xbase, 0, Dd);

  // Steady state: two K-steps per trip, buffers alternate roles.
  int tk = 0;
  for (; tk + 3 < nk; tk += 2) {
    a1 = gemm_load_a(ap, tk + 1);
    gemm_load_b(b1, xbase, tk + 1, Dd);
    gemm_wmma4(acc, a0, b0);

    a0 = gemm_load_a(ap, tk + 2);
    gemm_load_b(b0, xbase, tk + 2, Dd);
    gemm_wmma4(acc, a1, b1);
  }

  // Tail: last pair (tk, tk+1).
  a1 = gemm_load_a(ap, tk + 1);
  gemm_load_b(b1, xbase, tk + 1, Dd);
  gemm_wmma4(acc, a0, b0);
  gemm_wmma4(acc, a1, b1);

  float* __restrict__ ob = Out + (size_t)b * E * LL;
#pragma unroll
  for (int t = 0; t < NT; ++t) {
    const int col = l0 + t * 16 + l16;
#pragma unroll
    for (int r = 0; r < 8; ++r) {
      const int e = te * 16 + r + half * 8;
      if (e < E) ob[(size_t)e * LL + col] = acc[t][r];
    }
  }
}

// ---------------------------------------------------------------------------
// Kernel 3: causal depthwise conv1d (K=4) + SiLU.
// Emits f32 (B, D, L) for the scan and f16 (B, L, D) for the W_x WMMA GEMM.
// ---------------------------------------------------------------------------
__global__ __launch_bounds__(256) void conv_silu_kernel(
    const float* __restrict__ x,   // (B, DI, L)
    const float* __restrict__ cw,  // (DI, 4)
    const float* __restrict__ cb,  // (DI)
    float* __restrict__ xc,        // (B, DI, L) f32
    _Float16* __restrict__ xch)    // (B, L, DI) f16
{
  const size_t idx = (size_t)blockIdx.x * blockDim.x + threadIdx.x;
  if (idx >= (size_t)BB * DI * LL) return;
  const int l = (int)(idx % LL);
  const int d = (int)((idx / LL) % DI);
  const int b = (int)(idx / ((size_t)LL * DI));

  const float* xr = x + ((size_t)b * DI + d) * LL;
  float acc = cb[d];
#pragma unroll
  for (int k = 0; k < 4; ++k) {
    const int ls = l - 3 + k;
    if (ls >= 0) acc += cw[d * 4 + k] * xr[ls];
  }
  const float s = acc * sigmoidf_(acc);  // SiLU
  xc[((size_t)b * DI + d) * LL + l] = s;
  xch[((size_t)b * LL + l) * DI + d] = (_Float16)s;
}

// ---------------------------------------------------------------------------
// Kernel 4: fused selective scan + D-skip + SiLU(z) gating.
// One lane per inner channel d; h[16] lives in registers for the whole L loop.
// dt_low / B / C rows of xdbl are staged per 64-step chunk in LDS (shared
// across all 64 channels of the block).
// ---------------------------------------------------------------------------
#define SCAN_CH 64   // L chunk staged in LDS
__global__ __launch_bounds__(64) void scan_kernel(
    const float* __restrict__ xdbl,  // (B, 40, L)
    const float* __restrict__ xc,    // (B, DI, L)  post-conv activations
    const float* __restrict__ z,     // (B, DI, L)  gate pre-activation
    const float* __restrict__ Wdt,   // (DI, DTR)
    const float* __restrict__ bdt,   // (DI)
    const float* __restrict__ Alog,  // (DI, DS)
    const float* __restrict__ Dvec,  // (DI)
    _Float16* __restrict__ ygh)      // (B, L, DI) f16
{
  const int d = blockIdx.x * 64 + threadIdx.x;
  const int b = blockIdx.y;
  const int tid = threadIdx.x;

  // Per-channel constants in registers.
  float A_[DS], h[DS], wdt[DTR];
#pragma unroll
  for (int n = 0; n < DS; ++n) { A_[n] = -__expf(Alog[d * DS + n]); h[n] = 0.f; }
#pragma unroll
  for (int r = 0; r < DTR; ++r) wdt[r] = Wdt[d * DTR + r];
  const float bdt_d = bdt[d];
  const float Dd    = Dvec[d];

  __shared__ float sdt[DTR][SCAN_CH];
  __shared__ float sB[DS][SCAN_CH];
  __shared__ float sC[DS][SCAN_CH];

  const float* xdb = xdbl + (size_t)b * XROWS * LL;
  const float* xcd = xc + ((size_t)b * DI + d) * LL;
  const float* zd  = z  + ((size_t)b * DI + d) * LL;
  _Float16* yout   = ygh + (size_t)b * LL * DI + d;

  for (int lc = 0; lc < LL; lc += SCAN_CH) {
    __syncthreads();
    // Cooperative stage of xdbl[:, lc:lc+64] (40 rows) into LDS.
    for (int t = tid; t < XROWS * SCAN_CH; t += 64) {
      const int e = t >> 6;          // row 0..39
      const int j = t & (SCAN_CH - 1);
      const float v = xdb[(size_t)e * LL + lc + j];
      if (e < DTR)            sdt[e][j] = v;
      else if (e < DTR + DS)  sB[e - DTR][j] = v;
      else                    sC[e - DTR - DS][j] = v;
    }
    __syncthreads();

    for (int j = 0; j < SCAN_CH; ++j) {
      const int l = lc + j;
      // delta = softplus(W_dt @ dt_low + b_dt)
      float t = bdt_d;
#pragma unroll
      for (int r = 0; r < DTR; ++r) t += wdt[r] * sdt[r][j];
      const float delta = (t > 20.f) ? t : log1pf(__expf(t));

      const float xv = xcd[l];
      const float dx = delta * xv;

      float y = 0.f;
#pragma unroll
      for (int n = 0; n < DS; ++n) {
        const float da = __expf(delta * A_[n]);    // dA
        h[n] = da * h[n] + dx * sB[n][j];          // h = dA*h + dB*u
        y += h[n] * sC[n][j];                      // y = <h, C>
      }
      y += Dd * xv;                                // D skip
      const float zv = zd[l];
      y *= zv * sigmoidf_(zv);                     // * SiLU(z)
      yout[(size_t)l * DI] = (_Float16)y;
    }
  }
}

// ---------------------------------------------------------------------------
// Host-side launch
// ---------------------------------------------------------------------------
static inline unsigned ceil_div_u(size_t a, size_t b) { return (unsigned)((a + b - 1) / b); }

static void pack_w(const float* W, _Float16* Apack, int E, int Dd, hipStream_t stream) {
  const int Ep = (E + 15) & ~15;
  const size_t total = (size_t)Ep * Dd;
  pack_w_kernel<<<dim3(ceil_div_u(total, 256)), dim3(256), 0, stream>>>(W, Apack, E, Dd);
}

static void run_branch(const float* src, const float* nw, const float* nb,
                       const float* Win, const float* Wz,
                       const float* cw, const float* cb,
                       const float* Wx, const float* Wdt, const float* bdt,
                       const float* Alog, const float* Dvec, const float* Wout,
                       float* out,
                       _Float16* x3h, float* x, float* z, float* xc,
                       _Float16* xch, float* xdbl, _Float16* ygh,
                       _Float16* wpk_in, _Float16* wpk_z, _Float16* wpk_x, _Float16* wpk_out,
                       hipStream_t stream)
{
  // 0) pack the branch's weight matrices into WMMA A-fragment layout (f16)
  pack_w(Win,  wpk_in,  DI,    CC, stream);
  pack_w(Wz,   wpk_z,   DI,    CC, stream);
  pack_w(Wx,   wpk_x,   XROWS, DI, stream);
  pack_w(Wout, wpk_out, CC,    DI, stream);

  // 1) layernorm -> f16 (B, L, C)
  ln_f16_kernel<<<dim3(LL, BB), dim3(CC), 0, stream>>>(src, nw, nb, x3h);

  // 2) in-projections: x = W_in @ x3^T, z = W_z @ x3^T   (E=256, Dd=128)
  wmma_gemm_packed_kernel<<<dim3(LL / (16 * NT), DI / 16, BB), dim3(32), 0, stream>>>(wpk_in, x3h, x, DI, CC);
  wmma_gemm_packed_kernel<<<dim3(LL / (16 * NT), DI / 16, BB), dim3(32), 0, stream>>>(wpk_z,  x3h, z, DI, CC);

  // 3) causal conv1d + SiLU
  {
    const size_t n = (size_t)BB * DI * LL;
    conv_silu_kernel<<<dim3(ceil_div_u(n, 256)), dim3(256), 0, stream>>>(x, cw, cb, xc, xch);
  }

  // 4) x-projection: xdbl = W_x @ xconv   (E=40 -> 48 padded rows, Dd=256)
  wmma_gemm_packed_kernel<<<dim3(LL / (16 * NT), (XROWS + 15) / 16, BB), dim3(32), 0, stream>>>(wpk_x, xch, xdbl, XROWS, DI);

  // 5) fused selective scan + gating -> f16 (B, L, DI)
  scan_kernel<<<dim3(DI / 64, BB), dim3(64), 0, stream>>>(xdbl, xc, z, Wdt, bdt, Alog, Dvec, ygh);

  // 6) out-projection: out = W_out @ y_gated   (E=128, Dd=256) -> (B, C, L) == (B,C,H,W)
  wmma_gemm_packed_kernel<<<dim3(LL / (16 * NT), CC / 16, BB), dim3(32), 0, stream>>>(wpk_out, ygh, out, CC, DI);
}

extern "C" void kernel_launch(void* const* d_in, const int* in_sizes, int n_in,
                              void* d_out, int out_size, void* d_ws, size_t ws_size,
                              hipStream_t stream) {
  (void)in_sizes; (void)n_in; (void)out_size;

  const float* pan   = (const float*)d_in[0];
  const float* ms    = (const float*)d_in[1];
  const float* nwp   = (const float*)d_in[2];
  const float* nbp   = (const float*)d_in[3];
  const float* nwm   = (const float*)d_in[4];
  const float* nbm   = (const float*)d_in[5];
  const float* Winp  = (const float*)d_in[6];
  const float* Winm  = (const float*)d_in[7];
  const float* Wzp   = (const float*)d_in[8];
  const float* Wzm   = (const float*)d_in[9];
  const float* cwp   = (const float*)d_in[10];
  const float* cbp   = (const float*)d_in[11];
  const float* cwm   = (const float*)d_in[12];
  const float* cbm   = (const float*)d_in[13];
  const float* Wxp   = (const float*)d_in[14];
  const float* Wxm   = (const float*)d_in[15];
  const float* Wdtp  = (const float*)d_in[16];
  const float* Wdtm  = (const float*)d_in[17];
  const float* bdtp  = (const float*)d_in[18];
  const float* bdtm  = (const float*)d_in[19];
  const float* Alog  = (const float*)d_in[20];
  const float* Dp    = (const float*)d_in[21];
  const float* Dm    = (const float*)d_in[22];
  const float* Woutp = (const float*)d_in[23];
  const float* Woutm = (const float*)d_in[24];

  float* out_pan = (float*)d_out;                               // (B, C, L)
  float* out_ms  = out_pan + (size_t)BB * CC * LL;

  // Workspace carve-up (reused by both branches; ~35.5 MB total).
  uintptr_t base = (uintptr_t)d_ws;
  size_t off = 0;
  auto carve = [&](size_t bytes) -> uintptr_t {
    uintptr_t p = base + off;
    off += (bytes + 255) & ~(size_t)255;
    return p;
  };
  _Float16* x3h  = (_Float16*)carve((size_t)BB * LL * CC * sizeof(_Float16));
  float*    x    = (float*)   carve((size_t)BB * DI * LL * sizeof(float));
  float*    z    = (float*)   carve((size_t)BB * DI * LL * sizeof(float));
  float*    xc   = (float*)   carve((size_t)BB * DI * LL * sizeof(float));
  _Float16* xch  = (_Float16*)carve((size_t)BB * LL * DI * sizeof(_Float16));
  float*    xdbl = (float*)   carve((size_t)BB * XROWS * LL * sizeof(float));
  _Float16* ygh  = (_Float16*)carve((size_t)BB * LL * DI * sizeof(_Float16));
  _Float16* wpk_in  = (_Float16*)carve((size_t)DI * CC * sizeof(_Float16));
  _Float16* wpk_z   = (_Float16*)carve((size_t)DI * CC * sizeof(_Float16));
  _Float16* wpk_x   = (_Float16*)carve((size_t)48 * DI * sizeof(_Float16));
  _Float16* wpk_out = (_Float16*)carve((size_t)CC * DI * sizeof(_Float16));
  if (off > ws_size) return;  // insufficient scratch; nothing safe to do

  run_branch(pan, nwp, nbp, Winp, Wzp, cwp, cbp, Wxp, Wdtp, bdtp, Alog, Dp, Woutp,
             out_pan, x3h, x, z, xc, xch, xdbl, ygh,
             wpk_in, wpk_z, wpk_x, wpk_out, stream);
  run_branch(ms,  nwm, nbm, Winm, Wzm, cwm, cbm, Wxm, Wdtm, bdtm, Alog, Dm, Woutm,
             out_ms,  x3h, x, z, xc, xch, xdbl, ygh,
             wpk_in, wpk_z, wpk_x, wpk_out, stream);
}
